// MultiHeadAttention_78640851190516
// MI455X (gfx1250) — matrix operand order repaired and checked by reading, not compiled, not run on hardware
//
#include <hip/hip_runtime.h>
#include <hip/hip_bf16.h>

// ---------------------------------------------------------------------------
// MI455X (gfx1250) multi-head attention forward.
// Matrix math on v_wmma_f32_16x16x32_bf16 (wave32); GEMM K-chunks staged into
// double-buffered LDS with GLOBAL_LOAD_ASYNC_TO_LDS_B128, pipelined on ASYNCcnt.
// ---------------------------------------------------------------------------

typedef __attribute__((ext_vector_type(16))) __bf16          v16bf;
typedef __attribute__((ext_vector_type(8)))  float           v8f;
typedef __attribute__((ext_vector_type(16))) unsigned short  v16u;
typedef int v4i32 __attribute__((vector_size(16)));          // builtin pointee type

#define HIDDEN   1024
#define HEADS    16
#define HEAD_DIM 64
#define SEQ      2048
#define BATCH    2
#define MROWS    (BATCH * SEQ)        // 4096
#define BH       (BATCH * HEADS)      // 32

#define AS1 __attribute__((address_space(1)))
#define AS3 __attribute__((address_space(3)))

#if defined(__HIP_DEVICE_COMPILE__) && defined(__has_builtin)
#if __has_builtin(__builtin_amdgcn_global_load_async_to_lds_b128) && \
    __has_builtin(__builtin_amdgcn_s_wait_asynccnt)
#define HAVE_ASYNC_LDS 1
#endif
#endif

__device__ __forceinline__ unsigned short f2bf(float f) {
    unsigned u = __builtin_bit_cast(unsigned, f);
    unsigned r = (u + 0x7FFFu + ((u >> 16) & 1u)) >> 16;   // round-to-nearest-even
    return (unsigned short)r;
}

__device__ __forceinline__ v8f wmma_bf16(v16u a, v16u b, v8f c) {
    return __builtin_amdgcn_wmma_f32_16x16x32_bf16(
        /*neg_a=*/false, __builtin_bit_cast(v16bf, a),
        /*neg_b=*/false, __builtin_bit_cast(v16bf, b),
        /*c_mod=*/(short)0, c, /*reuse_a=*/false, /*reuse_b=*/false);
}

// A fragment (16x32 bf16, MxK): lanes 0-15 rows, half = lane>>4; element pairs:
//   VGPR v<4 : K = half*8 + 2v + p ; v>=4 : K = 16 + half*8 + 2(v-4) + p
__device__ __forceinline__ v16u load_a_u16(const unsigned short* base, int row0,
                                           int pitch, int koff, int lane) {
    int m = row0 + (lane & 15);
    int half = lane >> 4;
    const unsigned short* row = base + (size_t)m * pitch + koff;
    v16u r;
#pragma unroll
    for (int v = 0; v < 8; ++v) {
        int k = (v < 4) ? (half * 8 + v * 2) : (16 + half * 8 + (v - 4) * 2);
        unsigned d = *(const unsigned*)(row + k);
        r[2 * v]     = (unsigned short)(d & 0xffffu);
        r[2 * v + 1] = (unsigned short)(d >> 16);
    }
    return r;
}

// Same A layout, fp32 source with on-the-fly bf16 conversion.
__device__ __forceinline__ v16u load_a_f32(const float* base, int row0,
                                           int pitch, int koff, int lane) {
    int m = row0 + (lane & 15);
    int half = lane >> 4;
    const float* row = base + (size_t)m * pitch + koff;
    v16u r;
#pragma unroll
    for (int v = 0; v < 8; ++v) {
        int k = (v < 4) ? (half * 8 + v * 2) : (16 + half * 8 + (v - 4) * 2);
        float2 f = *(const float2*)(row + k);
        r[2 * v]     = f2bf(f.x);
        r[2 * v + 1] = f2bf(f.y);
    }
    return r;
}

// B fragment (32x16 bf16, KxN), rows-of-N contiguous in K:
//   lane N = row0 + (lane&15); element j<8 -> K = half*8 + j ; j>=8 -> K = 16 + half*8 + (j-8).
__device__ __forceinline__ v16u load_b_u16(const unsigned short* base, int row0,
                                           int pitch, int koff, int lane) {
    int n = row0 + (lane & 15);
    int half = lane >> 4;
    const unsigned short* p = base + (size_t)n * pitch + koff + half * 8;
    v16u r;
#pragma unroll
    for (int t = 0; t < 8; ++t) {
        unsigned d = *(const unsigned*)(p + ((t < 4) ? 2 * t : 16 + 2 * (t - 4)));
        r[2 * t]     = (unsigned short)(d & 0xffffu);
        r[2 * t + 1] = (unsigned short)(d >> 16);
    }
    return r;
}

// ---------------------------------------------------------------------------
// Kernel 0: bulk fp32 -> bf16 conversion (4 elements / thread).
// ---------------------------------------------------------------------------
__global__ __launch_bounds__(256) void f32_to_bf16_kernel(
    const float* __restrict__ src, unsigned short* __restrict__ dst)
{
    size_t i = ((size_t)blockIdx.x * 256 + threadIdx.x) * 4;
    float4 f = *(const float4*)(src + i);
    unsigned w0 = (unsigned)f2bf(f.x) | ((unsigned)f2bf(f.y) << 16);
    unsigned w1 = (unsigned)f2bf(f.z) | ((unsigned)f2bf(f.w) << 16);
    *(uint2*)(dst + i) = make_uint2(w0, w1);
}

// ---------------------------------------------------------------------------
// Kernel 1: Y[4096,1024] = X[4096,1024](bf16) * W[1024,1024](bf16)^T + bias.
// Block tile 128x64; K chunks of 32 in double-buffered LDS; async prefetch of
// chunk i+1 overlaps the WMMAs of chunk i (wait ASYNCcnt<=3 drains only chunk i).
// ---------------------------------------------------------------------------
#define GK 32
#define APITCH 40   // ushorts; 80B rows (16B aligned)
#define BPITCH 40

__device__ __forceinline__ void stage_tiles(
    const unsigned short* __restrict__ X, const unsigned short* __restrict__ W,
    int mBase, int nBase, int kt,
    unsigned short* __restrict__ la, unsigned short* __restrict__ lb, int t)
{
#if defined(HAVE_ASYNC_LDS)
    int c0 = t * 2;
#pragma unroll
    for (int cc = 0; cc < 2; ++cc) {        // A tile: 128 rows x 4 chunks(16B)
        int c = c0 + cc, row = c >> 2, q = c & 3;
        const unsigned short* g = X + (size_t)(mBase + row) * 1024 + kt + q * 8;
        __builtin_amdgcn_global_load_async_to_lds_b128(
            (AS1 v4i32*)g, (AS3 v4i32*)(la + row * APITCH + q * 8), 0, 0);
    }
    {                                        // B tile: 64 rows x 4 chunks
        int row = t >> 2, q = t & 3;
        const unsigned short* g = W + (size_t)(nBase + row) * 1024 + kt + q * 8;
        __builtin_amdgcn_global_load_async_to_lds_b128(
            (AS1 v4i32*)g, (AS3 v4i32*)(lb + row * BPITCH + q * 8), 0, 0);
    }
#else
    int c0 = t * 2;
#pragma unroll
    for (int cc = 0; cc < 2; ++cc) {
        int c = c0 + cc, row = c >> 2, q = c & 3;
        uint4 u = *(const uint4*)(X + (size_t)(mBase + row) * 1024 + kt + q * 8);
        *(uint4*)(la + row * APITCH + q * 8) = u;
    }
    {
        int row = t >> 2, q = t & 3;
        uint4 u = *(const uint4*)(W + (size_t)(nBase + row) * 1024 + kt + q * 8);
        *(uint4*)(lb + row * BPITCH + q * 8) = u;
    }
#endif
}

__global__ __launch_bounds__(256) void gemm_bf16_kernel(
    const unsigned short* __restrict__ X, const unsigned short* __restrict__ W,
    const float* __restrict__ bias, float* __restrict__ Y)
{
    __shared__ __align__(16) unsigned short ldsA[2][128 * APITCH];
    __shared__ __align__(16) unsigned short ldsB[2][64 * BPITCH];

    const int t    = threadIdx.x;
    const int lane = t & 31;
    const int w    = t >> 5;
    const int mBase = blockIdx.y * 128;
    const int nBase = blockIdx.x * 64;
    const int wm = (w >> 1) * 32;
    const int wn = (w & 1) * 32;

    v8f acc[2][2];
#pragma unroll
    for (int i = 0; i < 2; ++i)
#pragma unroll
        for (int j = 0; j < 2; ++j) acc[i][j] = (v8f)0.0f;

    stage_tiles(X, W, mBase, nBase, 0, ldsA[0], ldsB[0], t);   // prologue: chunk 0

    for (int kt = 0; kt < 1024; kt += GK) {
        const int cur = (kt >> 5) & 1;
        if (kt + GK < 1024) {
            stage_tiles(X, W, mBase, nBase, kt + GK, ldsA[cur ^ 1], ldsB[cur ^ 1], t);
#if defined(HAVE_ASYNC_LDS)
            __builtin_amdgcn_s_wait_asynccnt(3);   // chunk `cur` landed; 3 in flight
#endif
        } else {
#if defined(HAVE_ASYNC_LDS)
            __builtin_amdgcn_s_wait_asynccnt(0);
#endif
        }
        __syncthreads();                            // all waves see chunk `cur`

        v16u a0 = load_a_u16(ldsA[cur], wm,      APITCH, 0, lane);
        v16u a1 = load_a_u16(ldsA[cur], wm + 16, APITCH, 0, lane);
        v16u b0 = load_b_u16(ldsB[cur], wn,      BPITCH, 0, lane);
        v16u b1 = load_b_u16(ldsB[cur], wn + 16, BPITCH, 0, lane);
        acc[0][0] = wmma_bf16(a0, b0, acc[0][0]);
        acc[0][1] = wmma_bf16(a0, b1, acc[0][1]);
        acc[1][0] = wmma_bf16(a1, b0, acc[1][0]);
        acc[1][1] = wmma_bf16(a1, b1, acc[1][1]);
        __syncthreads();                            // reads done before buffer reuse
    }

    const int rbase = (lane < 16) ? 0 : 8;
#pragma unroll
    for (int i = 0; i < 2; ++i)
#pragma unroll
        for (int j = 0; j < 2; ++j) {
            int n = nBase + wn + j * 16 + (lane & 15);
            float bv = bias[n];
            float* p = Y + (size_t)(mBase + wm + i * 16 + rbase) * 1024 + n;
#pragma unroll
            for (int r = 0; r < 8; ++r)
                p[(size_t)r * 1024] = acc[i][j][r] + bv;   // immediate offsets
        }
}

// ---------------------------------------------------------------------------
// Kernel 2: RoPE + repack [B,S,1024] fp32 -> [B,H,S,64] bf16 (Q,K rotated; V copy)
// ---------------------------------------------------------------------------
__global__ __launch_bounds__(256) void rope_pack_kernel(
    const float* __restrict__ Qp, const float* __restrict__ Kp,
    const float* __restrict__ Vp,
    unsigned short* __restrict__ Qr, unsigned short* __restrict__ Kr,
    unsigned short* __restrict__ Vr)
{
    int i  = blockIdx.x * 256 + threadIdx.x;
    int d2 = i & 31;
    int s  = (i >> 5) & 2047;
    int h  = (i >> 16) & 15;
    int b  = i >> 20;
    size_t src = ((size_t)(b * SEQ + s)) * HIDDEN + h * HEAD_DIM + 2 * d2;
    size_t dst = ((size_t)((b * HEADS + h) * SEQ + s)) * HEAD_DIM + 2 * d2;

    float inv = __expf(-(float)(2 * d2) * (9.210340371976184f / 64.0f)); // 10000^(-2d/64)
    float ang = (float)s * inv;
    float sn, c;
    __sincosf(ang, &sn, &c);

    float q1 = Qp[src], q2 = Qp[src + 1];
    Qr[dst]     = f2bf(q1 * c - q2 * sn);
    Qr[dst + 1] = f2bf(q1 * sn + q2 * c);
    float k1 = Kp[src], k2 = Kp[src + 1];
    Kr[dst]     = f2bf(k1 * c - k2 * sn);
    Kr[dst + 1] = f2bf(k1 * sn + k2 * c);
    Vr[dst]     = f2bf(Vp[src]);
    Vr[dst + 1] = f2bf(Vp[src + 1]);
}

// ---------------------------------------------------------------------------
// Kernel 3: Vt[bh][d][s] = Vr[bh][s][d]  (64x64 LDS tiles, both sides coalesced)
// ---------------------------------------------------------------------------
__global__ __launch_bounds__(256) void transpose_v_kernel(
    const unsigned short* __restrict__ Vr, unsigned short* __restrict__ Vt)
{
    __shared__ __align__(16) unsigned short tile[64 * 80];   // pitch 80 (160B)
    const int bh = blockIdx.y;
    const int s0 = blockIdx.x * 64;
    const int r  = threadIdx.x >> 2;          // 0..63
    const int cq = (threadIdx.x & 3) * 16;    // 0,16,32,48

    {   // load 64(s) x 64(d), 32B per thread
        const uint4* g = (const uint4*)(Vr + ((size_t)bh * SEQ + s0 + r) * 64 + cq);
        *(uint4*)(tile + r * 80 + cq)     = g[0];
        *(uint4*)(tile + r * 80 + cq + 8) = g[1];
    }
    __syncthreads();
    {   // store 64(d) x 64(s), 32B per thread
        const int d = r, sq = cq;
        unsigned wds[8];
#pragma unroll
        for (int j = 0; j < 8; ++j) {
            unsigned short lo = tile[(sq + 2 * j)     * 80 + d];
            unsigned short hi = tile[(sq + 2 * j + 1) * 80 + d];
            wds[j] = (unsigned)lo | ((unsigned)hi << 16);
        }
        uint4* g = (uint4*)(Vt + ((size_t)bh * HEAD_DIM + d) * SEQ + s0 + sq);
        g[0] = make_uint4(wds[0], wds[1], wds[2], wds[3]);
        g[1] = make_uint4(wds[4], wds[5], wds[6], wds[7]);
    }
}

// ---------------------------------------------------------------------------
// Kernel 4: causal scores = (Q K^T) * 0.125, -1e9 above diagonal.
// One wave per 16-row query tile; Q/K fragments straight from global (L2-hot).
// ---------------------------------------------------------------------------
__global__ __launch_bounds__(256) void attn_scores_kernel(
    const unsigned short* __restrict__ Qr, const unsigned short* __restrict__ Kr,
    float* __restrict__ Wt)
{
    const int bh   = blockIdx.y;
    const int lane = threadIdx.x & 31;
    const int w    = threadIdx.x >> 5;
    const int q0   = (blockIdx.x * 8 + w) * 16;

    const unsigned short* Qb = Qr + (size_t)bh * SEQ * HEAD_DIM;
    const unsigned short* Kb = Kr + (size_t)bh * SEQ * HEAD_DIM;
    float* Wb = Wt + (size_t)bh * SEQ * SEQ;

    v16u aq0 = load_a_u16(Qb, q0, HEAD_DIM, 0,  lane);
    v16u aq1 = load_a_u16(Qb, q0, HEAD_DIM, 32, lane);

    const int rbase = (lane < 16) ? 0 : 8;
    const int ktmax = q0 >> 4;
    for (int kt = 0; kt <= ktmax; ++kt) {
        int k0 = kt * 16;
        v16u b0 = load_b_u16(Kb, k0, HEAD_DIM, 0,  lane);
        v16u b1 = load_b_u16(Kb, k0, HEAD_DIM, 32, lane);
        v8f acc = (v8f)0.0f;
        acc = wmma_bf16(aq0, b0, acc);
        acc = wmma_bf16(aq1, b1, acc);

        int kk = k0 + (lane & 15);
        float* p = Wb + (size_t)(q0 + rbase) * SEQ + kk;
#pragma unroll
        for (int r = 0; r < 8; ++r) {
            int qq = q0 + rbase + r;
            p[(size_t)r * SEQ] = (kk <= qq) ? acc[r] * 0.125f : -1.0e9f;
        }
    }
}

// ---------------------------------------------------------------------------
// Kernel 5: in-place row softmax over first (q+1) entries; exact zeros after.
// One wave per row.
// ---------------------------------------------------------------------------
__global__ __launch_bounds__(256) void softmax_rows_kernel(float* __restrict__ Wt)
{
    const int lane = threadIdx.x & 31;
    const int w    = threadIdx.x >> 5;
    const size_t row = (size_t)blockIdx.x * 8 + w;
    const int q = (int)(row & (SEQ - 1));
    float* base = Wt + row * SEQ;
    const int valid = q + 1;

    float m = -3.0e38f;
    for (int j0 = 0; j0 < valid; j0 += 32) {
        int j = j0 + lane;
        float v = (j < valid) ? base[j] : -3.0e38f;
        m = fmaxf(m, v);
    }
#pragma unroll
    for (int o = 16; o > 0; o >>= 1) m = fmaxf(m, __shfl_xor(m, o, 32));

    float s = 0.0f;
    for (int j0 = 0; j0 < valid; j0 += 32) {
        int j = j0 + lane;
        if (j < valid) s += __expf(base[j] - m);
    }
#pragma unroll
    for (int o = 16; o > 0; o >>= 1) s += __shfl_xor(s, o, 32);

    float invs = 1.0f / s;
    for (int j0 = 0; j0 < SEQ; j0 += 32) {
        int j = j0 + lane;
        float v = (j < valid) ? __expf(base[j] - m) * invs : 0.0f;
        base[j] = v;
    }
}

// ---------------------------------------------------------------------------
// Kernel 6: attn_out(bf16) = weights @ V, via transposed V (no LDS, no barriers).
// 8 waves x 16 query rows per block; softmax zeros make the uniform K loop exact.
// Streams the 512MB weights tensor; prefetch one chunk ahead.
// ---------------------------------------------------------------------------
__global__ __launch_bounds__(256) void attn_v_kernel(
    const float* __restrict__ Wt, const unsigned short* __restrict__ Vt,
    unsigned short* __restrict__ AO)
{
    const int bh = blockIdx.x;
    const int qb = blockIdx.y;
    const int b = bh >> 4, h = bh & 15;
    const int lane = threadIdx.x & 31, w = threadIdx.x >> 5;
    const int q0 = qb * 128 + w * 16;

    const float* Wb = Wt + (size_t)bh * SEQ * SEQ;
    const unsigned short* Vtb = Vt + (size_t)bh * HEAD_DIM * SEQ;

    v8f acc[4];
#pragma unroll
    for (int i = 0; i < 4; ++i) acc[i] = (v8f)0.0f;

    const int nk = (qb + 1) * 4;
    for (int kt = 0; kt < nk; ++kt) {
        int k0 = kt * 32;
        if (kt + 1 < nk)   // hint: next fp32 weight chunk for this lane's row
            __builtin_prefetch(Wb + (size_t)(q0 + (lane & 15)) * SEQ + k0 + 32, 0, 1);
        v16u a = load_a_f32(Wb, q0, SEQ, k0, lane);
#pragma unroll
        for (int nt = 0; nt < 4; ++nt) {
            v16u bf = load_b_u16(Vtb, nt * 16, SEQ, k0, lane);
            acc[nt] = wmma_bf16(a, bf, acc[nt]);
        }
    }

    const int rbase = (lane < 16) ? 0 : 8;
#pragma unroll
    for (int nt = 0; nt < 4; ++nt) {
        int d = nt * 16 + (lane & 15);
        unsigned short* p = AO + (size_t)(b * SEQ + q0 + rbase) * HIDDEN + h * HEAD_DIM + d;
#pragma unroll
        for (int r = 0; r < 8; ++r)
            p[(size_t)r * HIDDEN] = f2bf(acc[nt][r]);
    }
}

// ---------------------------------------------------------------------------
// Launcher
// ---------------------------------------------------------------------------
extern "C" void kernel_launch(void* const* d_in, const int* in_sizes, int n_in,
                              void* d_out, int out_size, void* d_ws, size_t ws_size,
                              hipStream_t stream)
{
    const float* query = (const float*)d_in[0];
    const float* key   = (const float*)d_in[1];
    const float* value = (const float*)d_in[2];
    const float* Wq    = (const float*)d_in[3];
    const float* bq    = (const float*)d_in[4];
    const float* Wk    = (const float*)d_in[5];
    const float* bk    = (const float*)d_in[6];
    const float* Wv    = (const float*)d_in[7];
    const float* bv    = (const float*)d_in[8];
    const float* Wo    = (const float*)d_in[9];
    const float* bo    = (const float*)d_in[10];

    float* out     = (float*)d_out;
    float* weights = out + (size_t)MROWS * HIDDEN;

    const size_t projElems = (size_t)MROWS * HIDDEN;        // 4 M
    const size_t bhsdElems = (size_t)BH * SEQ * HEAD_DIM;   // 4 M
    const size_t wElems    = (size_t)HIDDEN * HIDDEN;       // 1 M

    float* qproj   = (float*)d_ws;
    float* kproj   = qproj + projElems;
    float* vproj   = kproj + projElems;
    unsigned short* Xq  = (unsigned short*)(vproj + projElems);
    unsigned short* Xk  = Xq + projElems;
    unsigned short* Xv  = Xk + projElems;
    unsigned short* Wqb = Xv + projElems;
    unsigned short* Wkb = Wqb + wElems;
    unsigned short* Wvb = Wkb + wElems;
    unsigned short* Wob = Wvb + wElems;
    unsigned short* Qr  = Wob + wElems;
    unsigned short* Kr  = Qr + bhsdElems;
    unsigned short* Vr  = Kr + bhsdElems;
    unsigned short* Vt  = Vr + bhsdElems;
    unsigned short* AOb = Vt + bhsdElems;

    // 0) fp32 -> bf16 (inputs + weights)
    f32_to_bf16_kernel<<<projElems / 1024, 256, 0, stream>>>(query, Xq);
    f32_to_bf16_kernel<<<projElems / 1024, 256, 0, stream>>>(key,   Xk);
    f32_to_bf16_kernel<<<projElems / 1024, 256, 0, stream>>>(value, Xv);
    f32_to_bf16_kernel<<<wElems   / 1024, 256, 0, stream>>>(Wq, Wqb);
    f32_to_bf16_kernel<<<wElems   / 1024, 256, 0, stream>>>(Wk, Wkb);
    f32_to_bf16_kernel<<<wElems   / 1024, 256, 0, stream>>>(Wv, Wvb);
    f32_to_bf16_kernel<<<wElems   / 1024, 256, 0, stream>>>(Wo, Wob);

    dim3 gGemm(HIDDEN / 64, MROWS / 128);                   // (16, 32)

    // 1) projections
    gemm_bf16_kernel<<<gGemm, 256, 0, stream>>>(Xq, Wqb, bq, qproj);
    gemm_bf16_kernel<<<gGemm, 256, 0, stream>>>(Xk, Wkb, bk, kproj);
    gemm_bf16_kernel<<<gGemm, 256, 0, stream>>>(Xv, Wvb, bv, vproj);

    // 2) RoPE + head-major repack, 3) V transpose
    rope_pack_kernel<<<(BH * SEQ * 32) / 256, 256, 0, stream>>>(qproj, kproj, vproj, Qr, Kr, Vr);
    transpose_v_kernel<<<dim3(SEQ / 64, BH), 256, 0, stream>>>(Vr, Vt);

    // 4) causal scores, 5) softmax (writes attention weights output)
    attn_scores_kernel<<<dim3(SEQ / (16 * 8), BH), 256, 0, stream>>>(Qr, Kr, weights);
    softmax_rows_kernel<<<(BH * SEQ) / 8, 256, 0, stream>>>(weights);

    // 6) P @ V -> bf16 attn_out
    attn_v_kernel<<<dim3(BH, SEQ / 128), 256, 0, stream>>>(weights, Vt, AOb);

    // 7) output projection
    gemm_bf16_kernel<<<gGemm, 256, 0, stream>>>(AOb, Wob, bo, out);
}
